// MultiHeadAttn_8083128451275
// MI455X (gfx1250) — matrix operand order repaired and checked by reading, not compiled, not run on hardware
//
#include <hip/hip_runtime.h>
#include <hip/hip_bf16.h>
#include <math.h>

// ---------------------------------------------------------------------------
// MultiHeadAttn (post-LN) for MI455X / gfx1250.
// bf16 WMMA (v_wmma_f32_16x16x32_bf16) + TDM (tensor_load_to_lds) staging.
// ---------------------------------------------------------------------------

typedef __attribute__((ext_vector_type(16))) __bf16 v16bf;
typedef __attribute__((ext_vector_type(8)))  float  v8f;
typedef unsigned int u32x4 __attribute__((ext_vector_type(4)));
typedef int          i32x8 __attribute__((ext_vector_type(8)));
typedef int          i32x4 __attribute__((ext_vector_type(4)));

#define D_MODEL 1024
#define N_HEAD  16
#define D_HEAD  64
#define SEQ     2048
#define BATCH   2
#define NROW    (BATCH * SEQ)
#define ATTN_SCALE 0.125f
#define NEG_INF (-1e30f)

union FragAB {
    v16bf          v;
    uint4          q[2];
    unsigned short s[16];
};

__device__ __forceinline__ unsigned short f2bf(float f) {
    union { float f; unsigned u; } c; c.f = f;
    unsigned u = c.u;
    u += 0x7FFFu + ((u >> 16) & 1u);   // round-to-nearest-even
    return (unsigned short)(u >> 16);
}

__device__ __forceinline__ v8f wmma_bf16(const FragAB& a, const FragAB& b, v8f c) {
    return __builtin_amdgcn_wmma_f32_16x16x32_bf16(
        false, a.v, false, b.v, (short)0, c, false, false);
}

// A-fragment (16x32): lane L<16 -> row L, K={0..7,16..23}; lane L+16 -> K={8..15,24..31}
__device__ __forceinline__ FragAB load_a(const unsigned short* __restrict__ base,
                                         int ld, int row, int k0, int lane) {
    FragAB f;
    const int kb = (lane < 16) ? 0 : 8;
    const unsigned short* p = base + (size_t)row * ld + k0 + kb;
    f.q[0] = *reinterpret_cast<const uint4*>(p);
    f.q[1] = *reinterpret_cast<const uint4*>(p + 16);
    return f;
}

// B-fragment (32x16): lane L<16 -> column n0+L%16, K={0..7,16..23}; lane L+16 -> K={8..15,24..31}
__device__ __forceinline__ FragAB load_b(const unsigned short* __restrict__ w,
                                         int ld, int n0, int k0, int lane) {
    FragAB f;
    const int n  = n0 + (lane & 15);
    const int ks = k0 + ((lane < 16) ? 0 : 8);
    const unsigned short* p = w + (size_t)n * ld + ks;
    f.q[0] = *reinterpret_cast<const uint4*>(p);
    f.q[1] = *reinterpret_cast<const uint4*>(p + 16);
    return f;
}

// ---------------------------------------------------------------------------
// Tensor Data Mover: 2D tile (rows x row_bytes) global -> LDS.
// D# built per CDNA5 ISA ch.8: group0 {count=1, lds_addr, global_addr, type=2},
// group1 {data_size=8B, tensor_dim = tile_dim = (row_bytes/8) x rows}.
// 6-arg builtin form (clang-23): (g0, g1, g2, g3, g_extra, cpol).
// ---------------------------------------------------------------------------
__device__ __forceinline__ void tdm_load_2d(unsigned lds_byte_addr,
                                            const void* gptr,
                                            unsigned row_bytes, unsigned rows) {
    const unsigned long long ga = (unsigned long long)gptr;
    const unsigned units = row_bytes >> 3;            // 8-byte elements per row
    u32x4 g0;
    g0[0] = 1u;                                       // count=1 (valid user D#)
    g0[1] = lds_byte_addr;                            // [63:32] lds_addr
    g0[2] = (unsigned)(ga & 0xFFFFFFFFu);             // global_addr[31:0]
    g0[3] = (unsigned)((ga >> 32) & 0x1FFFFFFu)       // global_addr[56:32]
          | (2u << 30);                               // type=2 ("image")
    i32x8 g1;
    g1[0] = (int)(3u << 16);                          // data_size=3 (8B), mask=0
    g1[1] = (int)(units << 16);                       // tensor_dim0[15:0] @ [31:16]
    g1[2] = (int)((units >> 16) | (rows << 16));      // dim0 hi | tensor_dim1 lo
    g1[3] = (int)(units << 16);                       // dim1 hi(0) | tile_dim0
    g1[4] = (int)rows;                                // tile_dim1 | tile_dim2=0
    g1[5] = (int)units;                               // tensor_dim0_stride[31:0]
    g1[6] = 0;                                        // stride0 hi | stride1 lo
    g1[7] = 0;                                        // stride1 hi
    i32x4 z4 = {0, 0, 0, 0};                          // groups 2/3 unused (2D)
    i32x8 z8 = {0, 0, 0, 0, 0, 0, 0, 0};
    __builtin_amdgcn_tensor_load_to_lds(g0, g1, z4, z4, z8, 0);
}

// ---------------------------------------------------------------------------
// Kernel 0: fp32 -> bf16 conversion (grid-stride)
// ---------------------------------------------------------------------------
__global__ void cvt_bf16_kernel(const float* __restrict__ src,
                                unsigned short* __restrict__ dst, int n) {
    int i = blockIdx.x * blockDim.x + threadIdx.x;
    const int stride = gridDim.x * blockDim.x;
    for (; i < n; i += stride) dst[i] = f2bf(src[i]);
}

// ---------------------------------------------------------------------------
// Kernel 1: Q/K/V projection. 16-row block per workgroup (TDM-staged in LDS),
// 4 waves, each wave computes 16x32 output pairs with B double-buffering.
// ---------------------------------------------------------------------------
__global__ void __launch_bounds__(128)
qkv_kernel(const unsigned short* __restrict__ hb,
           const unsigned short* __restrict__ qwb,
           const unsigned short* __restrict__ kwb,
           const unsigned short* __restrict__ vwb,
           unsigned short* __restrict__ qb,
           unsigned short* __restrict__ kb,
           unsigned short* __restrict__ vtb) {
    __shared__ unsigned short hs[16 * D_MODEL];        // 32KB activation stage
    const int tid  = threadIdx.x;
    const int lane = tid & 31;
    const int wave = tid >> 5;
    const int m0   = blockIdx.x * 16;

    if (wave == 0) {
        const unsigned lds_addr = (unsigned)(unsigned long long)(void*)hs;
        tdm_load_2d(lds_addr, hb + (size_t)m0 * D_MODEL, D_MODEL * 2u, 16u);
        __builtin_amdgcn_s_wait_tensorcnt(0);
    }
    __syncthreads();

    // 96 tile-pairs (3 matrices x 32 pairs); uniform stride-4 over waves.
    for (int pp = wave; pp < 96; pp += 4) {
        const int mat = pp >> 5;
        const int n0  = (pp & 31) * 32;
        const unsigned short* w = (mat == 0) ? qwb : (mat == 1) ? kwb : vwb;

        v8f acc0 = {}, acc1 = {};
        FragAB b0 = load_b(w, D_MODEL, n0,      0, lane);
        FragAB b1 = load_b(w, D_MODEL, n0 + 16, 0, lane);
        for (int k0 = 0; k0 < D_MODEL; k0 += 32) {
            const int kn = (k0 + 32 < D_MODEL) ? k0 + 32 : 0;   // prefetch (wraps)
            FragAB nb0 = load_b(w, D_MODEL, n0,      kn, lane);
            FragAB nb1 = load_b(w, D_MODEL, n0 + 16, kn, lane);
            FragAB a;
            {
                const int kb2 = (lane < 16) ? 0 : 8;
                const unsigned short* p = hs + (lane & 15) * D_MODEL + k0 + kb2;
                a.q[0] = *reinterpret_cast<const uint4*>(p);
                a.q[1] = *reinterpret_cast<const uint4*>(p + 16);
            }
            acc0 = wmma_bf16(a, b0, acc0);
            acc1 = wmma_bf16(a, b1, acc1);
            b0 = nb0; b1 = nb1;
        }

        const int mo = (lane >> 4) * 8;
        for (int half = 0; half < 2; ++half) {
            const v8f& acc = half ? acc1 : acc0;
            const int  nc  = n0 + half * 16 + (lane & 15);
            if (mat < 2) {
                unsigned short* out = (mat == 0) ? qb : kb;
                for (int r = 0; r < 8; ++r)
                    out[(size_t)(m0 + mo + r) * D_MODEL + nc] = f2bf(acc[r]);
            } else {
                const int hh = nc >> 6, dc = nc & 63;
                for (int r = 0; r < 8; ++r) {
                    const int g = m0 + mo + r;
                    const int b = g >> 11, s = g & (SEQ - 1);
                    vtb[(((size_t)b * N_HEAD + hh) * D_HEAD + dc) * SEQ + s] =
                        f2bf(acc[r]);
                }
            }
        }
    }
}

// ---------------------------------------------------------------------------
// Kernel 2: flash attention. One wave per (b, head, 32-query block).
// 2 M-tiles share the K/V B-fragments; V is fetched before the softmax.
// ---------------------------------------------------------------------------
__global__ void __launch_bounds__(32)
attn_kernel(const unsigned short* __restrict__ qb,
            const unsigned short* __restrict__ kb,
            const unsigned short* __restrict__ vtb,
            const int* __restrict__ am,
            unsigned short* __restrict__ avb) {
    __shared__ float sc[32][32];
    __shared__ float aux[32];
    const int lane = threadIdx.x;
    const int h    = blockIdx.y;
    const int b    = blockIdx.z;
    const int m0   = blockIdx.x * 32;
    const size_t rowbase = (size_t)b * SEQ;

    // Q fragments: 2 M-tiles x 2 K-chunks
    FragAB aq[2][2];
    {
        const int kb2 = (lane < 16) ? 0 : 8;
        for (int m = 0; m < 2; ++m) {
            const unsigned short* p0 =
                qb + (rowbase + m0 + m * 16 + (lane & 15)) * (size_t)D_MODEL
                   + h * D_HEAD;
            for (int c = 0; c < 2; ++c) {
                const unsigned short* p = p0 + c * 32 + kb2;
                aq[m][c].q[0] = *reinterpret_cast<const uint4*>(p);
                aq[m][c].q[1] = *reinterpret_cast<const uint4*>(p + 16);
            }
        }
    }

    v8f acco[2][4];
    for (int m = 0; m < 2; ++m)
        for (int d = 0; d < 4; ++d) acco[m][d] = v8f{};
    float m_run = -INFINITY, l_run = 0.0f;   // lane owns softmax row `lane`

    for (int j0 = 0; j0 < SEQ; j0 += 32) {
        // --- K B-fragments (shared by both M-tiles) ---
        FragAB bk[2][2];                     // [score tile t][k-chunk c]
        for (int t = 0; t < 2; ++t)
            for (int c = 0; c < 2; ++c) {
                const int n  = j0 + t * 16 + (lane & 15);
                const int ks = c * 32 + ((lane < 16) ? 0 : 8);
                const unsigned short* p =
                    kb + (rowbase + n) * (size_t)D_MODEL + h * D_HEAD + ks;
                bk[t][c].q[0] = *reinterpret_cast<const uint4*>(p);
                bk[t][c].q[1] = *reinterpret_cast<const uint4*>(p + 16);
            }

        // --- V B-fragments prefetched now; consumed after softmax ---
        FragAB bv[4];
        for (int d = 0; d < 4; ++d) {
            const int dc = d * 16 + (lane & 15);
            const int js = j0 + ((lane < 16) ? 0 : 8);
            const unsigned short* p =
                vtb + (((size_t)b * N_HEAD + h) * D_HEAD + dc) * SEQ + js;
            bv[d].q[0] = *reinterpret_cast<const uint4*>(p);
            bv[d].q[1] = *reinterpret_cast<const uint4*>(p + 16);
        }

        // --- scores: 2 M-tiles x 2 N-tiles, K=64 ---
        const int  mo      = (lane >> 4) * 8;
        const bool masked0 = (am[b * SEQ + j0 + (lane & 15)] == 0);
        const bool masked1 = (am[b * SEQ + j0 + 16 + (lane & 15)] == 0);
        for (int m = 0; m < 2; ++m)
            for (int t = 0; t < 2; ++t) {
                v8f s = {};
                s = wmma_bf16(aq[m][0], bk[t][0], s);
                s = wmma_bf16(aq[m][1], bk[t][1], s);
                const bool masked = t ? masked1 : masked0;
                for (int r = 0; r < 8; ++r)
                    sc[m * 16 + mo + r][t * 16 + (lane & 15)] =
                        masked ? NEG_INF : s[r] * ATTN_SCALE;
            }
        __syncthreads();

        // --- online softmax: each of 32 lanes owns one row ---
        float alpha;
        {
            float mx = m_run;
            for (int j = 0; j < 32; ++j) mx = fmaxf(mx, sc[lane][j]);
            alpha = __expf(m_run - mx);
            float ps = 0.0f;
            for (int j = 0; j < 32; ++j) {
                const float p = __expf(sc[lane][j] - mx);
                sc[lane][j] = p;
                ps += p;
            }
            l_run = l_run * alpha + ps;
            m_run = mx;
            aux[lane] = alpha;
        }
        __syncthreads();

        // --- rescale O accumulators ---
        for (int m = 0; m < 2; ++m)
            for (int d = 0; d < 4; ++d)
                for (int r = 0; r < 8; ++r)
                    acco[m][d][r] *= aux[m * 16 + mo + r];

        // --- repack P into bf16 A-fragments ---
        FragAB ap[2];
        {
            const int kb2 = (lane < 16) ? 0 : 8;
            for (int m = 0; m < 2; ++m) {
                const int mrow = m * 16 + (lane & 15);
                for (int e = 0; e < 8; ++e) {
                    ap[m].s[e]     = f2bf(sc[mrow][kb2 + e]);
                    ap[m].s[e + 8] = f2bf(sc[mrow][16 + kb2 + e]);
                }
            }
        }
        __syncthreads();

        // --- O(32x64) += P(32x32) * V(32x64): 8 WMMAs ---
        for (int m = 0; m < 2; ++m)
            for (int d = 0; d < 4; ++d)
                acco[m][d] = wmma_bf16(ap[m], bv[d], acco[m][d]);
    }

    __syncthreads();
    aux[lane] = 1.0f / l_run;
    __syncthreads();

    {
        const int mo = (lane >> 4) * 8;
        for (int m = 0; m < 2; ++m)
            for (int d = 0; d < 4; ++d) {
                const int dc = d * 16 + (lane & 15);
                for (int r = 0; r < 8; ++r) {
                    const float v = acco[m][d][r] * aux[m * 16 + mo + r];
                    avb[(rowbase + m0 + m * 16 + mo + r) * (size_t)D_MODEL
                        + h * D_HEAD + dc] = f2bf(v);
                }
            }
    }
}

// ---------------------------------------------------------------------------
// Kernel 3: O projection + residual + LayerNorm (post-LN).
// ---------------------------------------------------------------------------
__global__ void __launch_bounds__(128)
oproj_ln_kernel(const unsigned short* __restrict__ avb,
                const unsigned short* __restrict__ owb,
                const float* __restrict__ hin,
                const float* __restrict__ lng,
                const float* __restrict__ lnb,
                float* __restrict__ out) {
    __shared__ float ob[16][D_MODEL];                  // 64KB pre-LN staging
    const int tid  = threadIdx.x;
    const int lane = tid & 31;
    const int wave = tid >> 5;
    const int m0   = blockIdx.x * 16;

    for (int pp = wave; pp < 32; pp += 4) {            // 32 column tile-pairs
        const int n0 = pp * 32;
        v8f acc0 = {}, acc1 = {};
        FragAB b0 = load_b(owb, D_MODEL, n0,      0, lane);
        FragAB b1 = load_b(owb, D_MODEL, n0 + 16, 0, lane);
        for (int k0 = 0; k0 < D_MODEL; k0 += 32) {
            const int kn = (k0 + 32 < D_MODEL) ? k0 + 32 : 0;
            FragAB nb0 = load_b(owb, D_MODEL, n0,      kn, lane);
            FragAB nb1 = load_b(owb, D_MODEL, n0 + 16, kn, lane);
            FragAB a   = load_a(avb, D_MODEL, m0 + (lane & 15), k0, lane);
            acc0 = wmma_bf16(a, b0, acc0);
            acc1 = wmma_bf16(a, b1, acc1);
            b0 = nb0; b1 = nb1;
        }
        const int mo = (lane >> 4) * 8;
        for (int half = 0; half < 2; ++half) {
            const v8f& acc = half ? acc1 : acc0;
            const int  nc  = n0 + half * 16 + (lane & 15);
            for (int r = 0; r < 8; ++r) {
                const int m = mo + r;
                ob[m][nc] = acc[r] + hin[(size_t)(m0 + m) * D_MODEL + nc];
            }
        }
    }
    __syncthreads();

    if (tid < 16) {
        float s = 0.0f, s2 = 0.0f;
        for (int c = 0; c < D_MODEL; ++c) {
            const float v = ob[tid][c];
            s += v; s2 += v * v;
        }
        const float mu  = s * (1.0f / D_MODEL);
        const float var = s2 * (1.0f / D_MODEL) - mu * mu;
        const float rs  = rsqrtf(var + 1e-5f);
        float* o = out + (size_t)(m0 + tid) * D_MODEL;
        for (int c = 0; c < D_MODEL; ++c)
            o[c] = (ob[tid][c] - mu) * rs * lng[c] + lnb[c];
    }
}

// ---------------------------------------------------------------------------
// Launcher
// ---------------------------------------------------------------------------
extern "C" void kernel_launch(void* const* d_in, const int* in_sizes, int n_in,
                              void* d_out, int out_size, void* d_ws, size_t ws_size,
                              hipStream_t stream) {
    (void)in_sizes; (void)n_in; (void)out_size; (void)ws_size;
    const float* h  = (const float*)d_in[0];
    const int*   am = (const int*)  d_in[1];
    const float* qw = (const float*)d_in[2];
    const float* kw = (const float*)d_in[3];
    const float* vw = (const float*)d_in[4];
    const float* ow = (const float*)d_in[5];
    const float* lg = (const float*)d_in[6];
    const float* lb = (const float*)d_in[7];
    float* out = (float*)d_out;

    const size_t HB = (size_t)NROW * D_MODEL;
    const size_t WB = (size_t)D_MODEL * D_MODEL;
    char* ws = (char*)d_ws;
    unsigned short* hb   = (unsigned short*)ws;        // 8MB (reused as avb)
    unsigned short* qwb  = (unsigned short*)(ws + HB * 2);
    unsigned short* kwb  = qwb + WB;
    unsigned short* vwb  = kwb + WB;
    unsigned short* owb  = vwb + WB;
    unsigned short* qb   = owb + WB;
    unsigned short* kbuf = qb + HB;
    unsigned short* vtb  = kbuf + HB;
    unsigned short* avb  = hb;                         // alias: hb dead after QKV

    cvt_bf16_kernel<<<1024, 256, 0, stream>>>(h,  hb,  (int)HB);
    cvt_bf16_kernel<<<256,  256, 0, stream>>>(qw, qwb, (int)WB);
    cvt_bf16_kernel<<<256,  256, 0, stream>>>(kw, kwb, (int)WB);
    cvt_bf16_kernel<<<256,  256, 0, stream>>>(vw, vwb, (int)WB);
    cvt_bf16_kernel<<<256,  256, 0, stream>>>(ow, owb, (int)WB);

    qkv_kernel<<<NROW / 16, 128, 0, stream>>>(hb, qwb, kwb, vwb, qb, kbuf, vtb);

    attn_kernel<<<dim3(SEQ / 32, N_HEAD, BATCH), 32, 0, stream>>>(
        qb, kbuf, vtb, am, avb);

    oproj_ln_kernel<<<NROW / 16, 128, 0, stream>>>(avb, owb, h, lg, lb, out);
}
